// MultiHeadAttention_7524782702820
// MI455X (gfx1250) — matrix operand order repaired and checked by reading, not compiled, not run on hardware
//
#include <hip/hip_runtime.h>
#include <hip/hip_bf16.h>

// ---------------------------------------------------------------------------
// MHA forward for MI455X (gfx1250, wave32, WMMA).
// Compute-bound (~500 FLOP/byte) -> f16 WMMA 16x16x32 with f32 accumulation
// (8x the per-instruction math of the f32 16x16x4 path).
// ---------------------------------------------------------------------------

typedef __attribute__((ext_vector_type(16))) _Float16 v16h;
typedef __attribute__((ext_vector_type(4)))  _Float16 v4h;
typedef __attribute__((ext_vector_type(8)))  float    v8f;
typedef __attribute__((ext_vector_type(4)))  float    v4f;

union Frag { v16h h; v4f f[2]; };     // one WMMA A/B fragment = two 16B loads
union H8   { v4f f; _Float16 h[8]; };

#define B_  2
#define S_  2048
#define D_  1024
#define H_  16
#define DH_ 64
#define M_  (B_*S_)   // 4096 rows

static __device__ inline v8f wmma_f16(v16h a, v16h b, v8f c) {
  // (neg_a, A, neg_b, B, c_mod, C, reuse_a, reuse_b)
  return __builtin_amdgcn_wmma_f32_16x16x32_f16(false, a, false, b, (short)0, c,
                                                false, false);
}

// ---------------------------------------------------------------------------
// 1) x (f32) -> xh (f16), straight layout [B*S][D]
// ---------------------------------------------------------------------------
__global__ void __launch_bounds__(256)
cvt_x_kernel(const float* __restrict__ x, _Float16* __restrict__ xh) {
  size_t i = ((size_t)blockIdx.x * 256 + threadIdx.x) * 4;
  v4f v = *(const v4f*)(x + i);
  v4h o = {(_Float16)v.x, (_Float16)v.y, (_Float16)v.z, (_Float16)v.w};
  *(v4h*)(xh + i) = o;
}

// ---------------------------------------------------------------------------
// 2) W[k][n] (f32) -> Wt[n][k] (f16), so B-fragments are contiguous per lane.
// ---------------------------------------------------------------------------
__global__ void __launch_bounds__(256)
wtrans_kernel(const float* __restrict__ Wq, const float* __restrict__ Wk,
              const float* __restrict__ Wv, _Float16* __restrict__ wth) {
  __shared__ float tile[32][33];
  const int wsel = blockIdx.z;
  const float* W = (wsel == 0) ? Wq : ((wsel == 1) ? Wk : Wv);
  const int n0 = blockIdx.x * 32, k0 = blockIdx.y * 32;
  const int tx = threadIdx.x & 31, ty = threadIdx.x >> 5;   // 32 x 8
  for (int i = 0; i < 4; i++)
    tile[ty + i * 8][tx] = W[(size_t)(k0 + ty + i * 8) * D_ + n0 + tx];
  __syncthreads();
  _Float16* dst = wth + (size_t)wsel * D_ * D_;
  for (int i = 0; i < 4; i++)
    dst[(size_t)(n0 + ty + i * 8) * D_ + k0 + tx] = (_Float16)tile[tx][ty + i * 8];
}

// ---------------------------------------------------------------------------
// 3) QKV projection GEMM: out[w][b][h][s][dh] (f16) = x @ W + bias
//    Workgroup: 128(M) x 64(N) tile, 8 waves of 32x32, K-chunks of 32.
// ---------------------------------------------------------------------------
__global__ void __launch_bounds__(256)
qkv_gemm_kernel(const _Float16* __restrict__ xh, const _Float16* __restrict__ wth,
                const float* __restrict__ bq, const float* __restrict__ bk,
                const float* __restrict__ bv, _Float16* __restrict__ qkvh) {
  constexpr int LS = 40;                         // padded row stride (halfs)
  __shared__ __align__(16) _Float16 ldsx[128 * LS];
  __shared__ __align__(16) _Float16 ldsw[64 * LS];

  const int m0   = blockIdx.x * 128;
  const int wsel = blockIdx.y >> 4;              // 0=Q 1=K 2=V
  const int n0   = (blockIdx.y & 15) * 64;
  const int tid  = threadIdx.x;
  const int lane = tid & 31, wave = tid >> 5;
  const int wm = wave >> 1, wn = wave & 1;       // 4x2 wave grid
  const int lm = lane & 15, g = lane >> 4;
  const _Float16* Wbase = wth + (size_t)wsel * (D_ * D_);

  v8f acc[2][2] = {};
  const int srow = tid >> 1, sseg = tid & 1;

  for (int kc = 0; kc < D_; kc += 32) {
    __syncthreads();
    {  // stage 128x32 halfs of x
      const v4f* src = (const v4f*)(xh + (size_t)(m0 + srow) * D_ + kc + sseg * 16);
      v4f* dst = (v4f*)(ldsx + srow * LS + sseg * 16);
      dst[0] = src[0]; dst[1] = src[1];
    }
    if (tid < 128) {  // stage 64x32 halfs of Wt
      const v4f* src = (const v4f*)(Wbase + (size_t)(n0 + srow) * D_ + kc + sseg * 16);
      v4f* dst = (v4f*)(ldsw + srow * LS + sseg * 16);
      dst[0] = src[0]; dst[1] = src[1];
    }
    if (kc + 32 < D_)  // global_prefetch_b8 for next K slab
      __builtin_prefetch(xh + (size_t)(m0 + srow) * D_ + kc + 32 + sseg * 16, 0, 3);
    __syncthreads();

    v16h afr[2], bfr[2];
    for (int mi = 0; mi < 2; mi++) {  // A frag: halfs 0-7 @ K=g*8, 8-15 @ 16+g*8
      const _Float16* p = ldsx + (wm * 32 + mi * 16 + lm) * LS;
      Frag u; u.f[0] = *(const v4f*)(p + g * 8); u.f[1] = *(const v4f*)(p + 16 + g * 8);
      afr[mi] = u.h;
    }
    for (int ni = 0; ni < 2; ni++) {  // B frag: halfs 0-7 @ K=g*8, 8-15 @ 16+g*8
      const _Float16* p = ldsw + (wn * 32 + ni * 16 + lm) * LS + g * 8;
      Frag u; u.f[0] = *(const v4f*)(p); u.f[1] = *(const v4f*)(p + 16);
      bfr[ni] = u.h;
    }
    for (int mi = 0; mi < 2; mi++)
      for (int ni = 0; ni < 2; ni++)
        acc[mi][ni] = wmma_f16(afr[mi], bfr[ni], acc[mi][ni]);
  }

  // Epilogue: bias add, scatter as f16 into [w][b][h][s][dh]
  const float* bias = (wsel == 0) ? bq : ((wsel == 1) ? bk : bv);
  for (int mi = 0; mi < 2; mi++)
    for (int ni = 0; ni < 2; ni++) {
      const int n  = n0 + wn * 32 + ni * 16 + lm;
      const float bb = bias[n];
      const int hh = n >> 6, dh = n & 63;
      for (int r = 0; r < 8; r++) {           // C layout: row = r + 8*g
        const int m  = m0 + wm * 32 + mi * 16 + r + 8 * g;
        const int bi = m >> 11, ss = m & (S_ - 1);
        const size_t off = ((((size_t)wsel * B_ + bi) * H_ + hh) * S_ + ss) * DH_ + dh;
        qkvh[off] = (_Float16)(acc[mi][ni][r] + bb);
      }
    }
}

// ---------------------------------------------------------------------------
// 4) Flash attention per (b, h, 64-query tile). 4 waves x 16 queries.
//    K staged [key][d] (B-frag for Q*K^T), V staged transposed [d][key]
//    (B-frag for P*V). Online softmax with shfl_xor row reductions.
// ---------------------------------------------------------------------------
__global__ void __launch_bounds__(128)
attn_kernel(const _Float16* __restrict__ qkvh, float* __restrict__ out) {
  constexpr int KP = 72;  // padded LDS row stride (halfs), multiple of 8
  __shared__ __align__(16) _Float16 Kt[64 * KP];
  __shared__ __align__(16) _Float16 Vt[64 * KP];
  __shared__ __align__(16) _Float16 Pt[4][16 * KP];

  const int q0 = blockIdx.x * 64;
  const int h  = blockIdx.y, b = blockIdx.z;
  const int tid = threadIdx.x, lane = tid & 31, wv = tid >> 5;
  const int lm = lane & 15, g = lane >> 4;

  const size_t bh = ((size_t)b * H_ + h) * S_ * DH_;
  const size_t one = (size_t)B_ * H_ * S_ * DH_;
  const _Float16* Qp = qkvh + bh;
  const _Float16* Kp = qkvh + one + bh;
  const _Float16* Vp = qkvh + 2 * one + bh;

  // Q fragments (16 queries x 64 d = 2 K-chunks), direct from global
  v16h qfr[2];
  {
    const _Float16* qrow = Qp + (size_t)(q0 + wv * 16 + lm) * DH_;
    for (int kc = 0; kc < 2; kc++) {
      Frag u;
      u.f[0] = *(const v4f*)(qrow + kc * 32 + g * 8);
      u.f[1] = *(const v4f*)(qrow + kc * 32 + 16 + g * 8);
      qfr[kc] = u.h;
    }
  }

  float mrow[8], lrow[8];
  v8f oacc[4] = {};
  for (int r = 0; r < 8; r++) { mrow[r] = -__builtin_inff(); lrow[r] = 0.f; }

  const int nkb = blockIdx.x + 1;  // causal: only key blocks <= query block
  for (int kb = 0; kb < nkb; kb++) {
    __syncthreads();
    {  // stage K block: [key][d]
      const int key = tid >> 1, seg = tid & 1;
      const v4f* src = (const v4f*)(Kp + (size_t)(kb * 64 + key) * DH_ + seg * 32);
      v4f* dst = (v4f*)(Kt + key * KP + seg * 32);
      dst[0] = src[0]; dst[1] = src[1];
      dst[2] = src[2]; dst[3] = src[3];
    }
    {  // stage V block transposed: [d][key]
      const int key = tid & 63, dh2 = tid >> 6;  // d range dh2*32..+31
      const _Float16* src = Vp + (size_t)(kb * 64 + key) * DH_ + dh2 * 32;
      for (int j0 = 0; j0 < 32; j0 += 8) {
        H8 c; c.f = *(const v4f*)(src + j0);
        for (int j = 0; j < 8; j++) Vt[(dh2 * 32 + j0 + j) * KP + key] = c.h[j];
      }
    }
    __syncthreads();

    // scores: 16 queries x 64 keys  (2 K-chunks x 4 N-subtiles)
    v8f sc[4] = {};
    for (int kc = 0; kc < 2; kc++)
      for (int ns = 0; ns < 4; ns++) {
        const _Float16* p = Kt + (ns * 16 + lm) * KP + kc * 32 + g * 8;
        Frag u; u.f[0] = *(const v4f*)(p); u.f[1] = *(const v4f*)(p + 16);
        sc[ns] = wmma_f16(qfr[kc], u.h, sc[ns]);
      }

    // online softmax (C layout: row = r + 8*g, col = lane%16 within subtile)
    float alpha[8];
    const int qbase = q0 + wv * 16;
    for (int r = 0; r < 8; r++) {
      const int qg = qbase + r + 8 * g;
      float mx = -__builtin_inff();
      for (int ns = 0; ns < 4; ns++) {
        const int kg = kb * 64 + ns * 16 + lm;
        float s = sc[ns][r] * 0.125f;            // 1/sqrt(64)
        if (kg > qg) s = -__builtin_inff();      // causal mask
        sc[ns][r] = s;
        mx = fmaxf(mx, s);
      }
      for (int d2 = 1; d2 < 16; d2 <<= 1) mx = fmaxf(mx, __shfl_xor(mx, d2, 32));
      const float mnew = fmaxf(mrow[r], mx);
      alpha[r] = __expf(mrow[r] - mnew);
      float rs = 0.f;
      for (int ns = 0; ns < 4; ns++) {
        const float p = __expf(sc[ns][r] - mnew);
        sc[ns][r] = p;
        rs += p;
      }
      for (int d2 = 1; d2 < 16; d2 <<= 1) rs += __shfl_xor(rs, d2, 32);
      lrow[r] = lrow[r] * alpha[r] + rs;
      mrow[r] = mnew;
    }

    // P (C layout) -> LDS (f16) to reshape into A-fragment layout
    for (int ns = 0; ns < 4; ns++)
      for (int r = 0; r < 8; r++)
        Pt[wv][(r + 8 * g) * KP + ns * 16 + lm] = (_Float16)sc[ns][r];
    for (int ds = 0; ds < 4; ds++)
      for (int r = 0; r < 8; r++) oacc[ds][r] *= alpha[r];

    // O += P @ V   (2 K-chunks over keys x 4 d-subtiles)
    const _Float16* prow = &Pt[wv][lm * KP];
    for (int kc2 = 0; kc2 < 2; kc2++) {
      Frag ua;
      ua.f[0] = *(const v4f*)(prow + kc2 * 32 + g * 8);
      ua.f[1] = *(const v4f*)(prow + kc2 * 32 + 16 + g * 8);
      for (int ds = 0; ds < 4; ds++) {
        const _Float16* p = Vt + (ds * 16 + lm) * KP + kc2 * 32 + g * 8;
        Frag ub; ub.f[0] = *(const v4f*)(p); ub.f[1] = *(const v4f*)(p + 16);
        oacc[ds] = wmma_f16(ua.h, ub.h, oacc[ds]);
      }
    }
  }

  // epilogue: normalize, write f32 to out[b][s][h*64+d]
  for (int ds = 0; ds < 4; ds++)
    for (int r = 0; r < 8; r++) {
      const int m = q0 + wv * 16 + r + 8 * g;
      const int dcol = h * DH_ + ds * 16 + lm;
      out[((size_t)b * S_ + m) * D_ + dcol] = oacc[ds][r] / lrow[r];
    }
}

// ---------------------------------------------------------------------------
extern "C" void kernel_launch(void* const* d_in, const int* in_sizes, int n_in,
                              void* d_out, int out_size, void* d_ws, size_t ws_size,
                              hipStream_t stream) {
  (void)in_sizes; (void)n_in; (void)out_size; (void)ws_size;
  const float* x  = (const float*)d_in[0];
  const float* Wq = (const float*)d_in[1];
  const float* bq = (const float*)d_in[2];
  const float* Wk = (const float*)d_in[3];
  const float* bk = (const float*)d_in[4];
  const float* Wv = (const float*)d_in[5];
  const float* bv = (const float*)d_in[6];
  float* out = (float*)d_out;

  // workspace: xh 8 MiB | Wt(f16) 6 MiB | qkv(f16) 24 MiB  (38 MiB total)
  char* ws = (char*)d_ws;
  _Float16* xh   = (_Float16*)(ws);
  _Float16* wth  = (_Float16*)(ws + ((size_t)8 << 20));
  _Float16* qkvh = (_Float16*)(ws + ((size_t)14 << 20));

  cvt_x_kernel<<<(M_ * D_) / (256 * 4), 256, 0, stream>>>(x, xh);
  wtrans_kernel<<<dim3(D_ / 32, D_ / 32, 3), 256, 0, stream>>>(Wq, Wk, Wv, wth);
  qkv_gemm_kernel<<<dim3(M_ / 128, 48), 256, 0, stream>>>(xh, wth, bq, bk, bv, qkvh);
  attn_kernel<<<dim3(S_ / 64, H_, B_), 128, 0, stream>>>(qkvh, out);
}
